// FirstToSpike_51840255262754
// MI455X (gfx1250) — matrix operand order repaired and checked by reading, not compiled, not run on hardware
//
#include <hip/hip_runtime.h>

#define DIN   2312
#define TT    300
#define WIN   50
#define HID   800
#define NCOUT 10
#define BATCH 128
#define RB    16            // batch rows per block
#define KT    73            // K tiles of 32 (K padded 2312 -> 2336)
#define NTILE 50            // HID tiles of 16
#define W1S_ELEMS (NTILE * KT * 512)   // pre-swizzled bf16 W1 elements
#define AS_ELEMS  (KT * 512)           // per-step A panel (16 x 2336 bf16)

typedef __attribute__((ext_vector_type(16))) __bf16 v16bf;
typedef __attribute__((ext_vector_type(8)))  float  v8f;

// ---------------------------------------------------------------------------
// Kernel 1: paste input clip into X output (bandwidth-bound streaming copy).
// ---------------------------------------------------------------------------
__global__ void paste_kernel(const float* __restrict__ input,
                             const float* __restrict__ Xin,
                             const int*   __restrict__ idx,
                             float* __restrict__ Xout,
                             long long total) {
  long long g = (long long)blockIdx.x * blockDim.x + threadIdx.x;
  if (g >= total) return;
  int t = (int)(g % TT);
  long long rem = g / TT;
  int d = (int)(rem % DIN);
  int i = (int)(rem / DIN);
  int s = t - idx[i];
  float v = (s >= 0 && s < WIN) ? input[((long long)i * DIN + d) * WIN + s]
                                : Xin[g];
  Xout[g] = v;
}

// ---------------------------------------------------------------------------
// Kernel 2: pre-swizzle W1 into the bf16 B-operand lane layout.
// offset = ((nt*KT + kt)*512) + lane*16 + j ; lane holds col n=nt*16+(L&15),
// K range kt*32 + (L>>4)*16 .. +16 contiguous.
// ---------------------------------------------------------------------------
__global__ void w1_swizzle_kernel(const float* __restrict__ W1,
                                  __bf16* __restrict__ w1s) {
  int g = blockIdx.x * blockDim.x + threadIdx.x;
  if (g >= W1S_ELEMS) return;
  int j  = g & 15;
  int l  = (g >> 4) & 31;
  int nk = g >> 9;
  int kt = nk % KT;
  int nt = nk / KT;
  int n  = nt * 16 + (l & 15);
  int k  = kt * 32 + ((l >> 4) & 1) * 16 + j;
  float v = (k < DIN) ? W1[(long long)n * DIN + k] : 0.0f;
  w1s[g] = (__bf16)v;
}

// ---------------------------------------------------------------------------
// Kernel 3: 300-step SNN scan. Block = 16 batch rows, 320 threads (10 waves).
// A-panel staged once per step into dynamic LDS in WMMA A-operand layout;
// B double-buffered in registers; h1m persists in WMMA accumulators.
// ---------------------------------------------------------------------------
extern __shared__ __align__(32) __bf16 aS[];   // [KT*512] dynamic LDS

__global__ __launch_bounds__(320)
void snn_scan_kernel(const float*  __restrict__ input,
                     const int*    __restrict__ idx,
                     const __bf16* __restrict__ w1s,
                     const float*  __restrict__ b1,
                     const float*  __restrict__ W2,
                     const float*  __restrict__ b2,
                     const float*  __restrict__ Wc,
                     const float*  __restrict__ bc,
                     float* __restrict__ rate_out) {
  __shared__ __bf16 h1sS[RB * HID];        // h1 spikes (0/1)
  __shared__ __bf16 W2s[NCOUT * HID];
  __shared__ __bf16 Wcs[HID + 4];
  __shared__ float  b2s[NCOUT];
  __shared__ float  bcs;
  __shared__ float  csS[RB];               // control spike carried into step
  __shared__ float  cmS[RB];
  __shared__ float  bgtS[RB];
  __shared__ float  actS[RB];              // row-active flag for current step
  __shared__ int    idxS[RB];
  __shared__ float  h2mS[RB * NCOUT];
  __shared__ float  h2sS[RB * NCOUT];
  __shared__ float  sum2S[RB * NCOUT];

  const int tid  = threadIdx.x;
  const int wid  = tid >> 5;       // wave id 0..9
  const int lane = tid & 31;
  const int r0   = blockIdx.x * RB;

  // ---- one-time staging ----
  for (int i = tid; i < NCOUT * HID; i += 320) W2s[i] = (__bf16)W2[i];
  for (int i = tid; i < HID + 4;    i += 320) Wcs[i] = (__bf16)Wc[i];
  if (tid < NCOUT) b2s[tid] = b2[tid];
  if (tid == 0)    bcs = bc[0];
  if (tid < RB) {
    csS[tid] = 0.0f; cmS[tid] = 0.0f; bgtS[tid] = 1.0f;
    int id = idx[r0 + tid];
    idxS[tid] = id;
    actS[tid] = (id == 0) ? 1.0f : 0.0f;   // t=0: gate forced 1, window@0
  }
  for (int i = tid; i < RB * NCOUT; i += 320) {
    h2mS[i] = 0.0f; h2sS[i] = 0.0f; sum2S[i] = 0.0f;
  }

  const int khalf = (lane >> 4) & 1;       // C-layout row-half select
  float b1reg[5];
  #pragma unroll
  for (int tl = 0; tl < 5; ++tl)
    b1reg[tl] = b1[wid * 80 + tl * 16 + (lane & 15)];

  const __bf16* wl = w1s + lane * 16;      // per-lane W1 base

  v8f acc[5];                              // persistent h1m (5 N-tiles)
  #pragma unroll
  for (int tl = 0; tl < 5; ++tl) acc[tl] = {};

  __syncthreads();

  #pragma unroll 1
  for (int t = 0; t < TT; ++t) {
    // -- phase A: h1m = 0.1*h1m*(1-h1s) + b1 (+ x_a @ W1^T via WMMA) --------
    #pragma unroll
    for (int tl = 0; tl < 5; ++tl) {
      v8f c;
      #pragma unroll
      for (int e = 0; e < 8; ++e) {
        float m = acc[tl][e];
        c[e] = ((m > 0.5f) ? 0.0f : m * 0.1f) + b1reg[tl];
      }
      acc[tl] = c;
    }

    bool myAct = (actS[lane & 15] != 0.0f);
    if (__any((int)myAct)) {               // x exactly 0 for all rows -> skip
      // stage gated A panel once per block, pre-swizzled for the A operand
      #pragma unroll 1
      for (int i = tid; i < AS_ELEMS; i += 320) {
        int j   = i & 15;
        int l   = (i >> 4) & 31;
        int kt  = i >> 9;
        int row = l & 15;
        int k   = kt * 32 + ((l >> 4) & 1) * 16 + j;
        float v = 0.0f;
        if (actS[row] != 0.0f && k < DIN) {
          int s = t - idxS[row];           // in [0,WIN) when active
          v = input[((long long)(r0 + row) * DIN + k) * WIN + s];
        }
        aS[i] = (__bf16)v;
      }
      __syncthreads();                     // A panel visible (uniform branch)

      // K loop: B double-buffered in registers, A from LDS, 5 WMMA per tile
      v16bf bb0[5], bb1[5];
      #pragma unroll
      for (int tl = 0; tl < 5; ++tl)
        bb0[tl] = *(const v16bf*)(wl + (long long)((wid * 5 + tl) * KT) * 512);

      #pragma unroll 1
      for (int kt = 0; kt < KT - 1; kt += 2) {
        v16bf a0 = *(const v16bf*)(aS + kt * 512 + lane * 16);
        #pragma unroll
        for (int tl = 0; tl < 5; ++tl)
          bb1[tl] = *(const v16bf*)(wl +
              (long long)((wid * 5 + tl) * KT + kt + 1) * 512);
        #pragma unroll
        for (int tl = 0; tl < 5; ++tl)
          acc[tl] = __builtin_amdgcn_wmma_f32_16x16x32_bf16(
              false, a0, false, bb0[tl], (short)0, acc[tl], false, false);

        v16bf a1 = *(const v16bf*)(aS + (kt + 1) * 512 + lane * 16);
        if (kt + 2 < KT) {
          #pragma unroll
          for (int tl = 0; tl < 5; ++tl)
            bb0[tl] = *(const v16bf*)(wl +
                (long long)((wid * 5 + tl) * KT + kt + 2) * 512);
        }
        #pragma unroll
        for (int tl = 0; tl < 5; ++tl)
          acc[tl] = __builtin_amdgcn_wmma_f32_16x16x32_bf16(
              false, a1, false, bb1[tl], (short)0, acc[tl], false, false);
      }
      { // tail tile kt = KT-1 (held in bb0)
        v16bf a0 = *(const v16bf*)(aS + (KT - 1) * 512 + lane * 16);
        #pragma unroll
        for (int tl = 0; tl < 5; ++tl)
          acc[tl] = __builtin_amdgcn_wmma_f32_16x16x32_bf16(
              false, a0, false, bb0[tl], (short)0, acc[tl], false, false);
      }
    }

    // spikes -> LDS (C/D layout: VGPR e = batch row e + 8*khalf, N = lane&15)
    #pragma unroll
    for (int tl = 0; tl < 5; ++tl) {
      int col = wid * 80 + tl * 16 + (lane & 15);
      #pragma unroll
      for (int e = 0; e < 8; ++e) {
        int rr = e + khalf * 8;
        h1sS[rr * HID + col] = (__bf16)((acc[tl][e] > 0.5f) ? 1.0f : 0.0f);
      }
    }

    __syncthreads();   // h1s visible; all actS/aS reads of this step done

    // -- phase B: tiny h2 / control matvecs + state updates -----------------
    if (tid < RB * NCOUT) {
      int rr = tid & 15, c = tid >> 4;
      float dot = b2s[c];
      const __bf16* hrow = &h1sS[rr * HID];
      const __bf16* wrow = &W2s[c * HID];
      #pragma unroll 4
      for (int h = 0; h < HID; ++h)
        dot += (float)hrow[h] * (float)wrow[h];
      float m  = h2mS[tid];
      float sp = h2sS[tid];
      m = m * 0.1f * (1.0f - sp) + dot;
      float ns = (m > 0.5f) ? 1.0f : 0.0f;
      h2mS[tid] = m; h2sS[tid] = ns; sum2S[tid] += ns;
    }
    if (tid < RB) {
      int rr = tid;
      float dot = bcs;
      const __bf16* hrow = &h1sS[rr * HID];
      #pragma unroll 4
      for (int h = 0; h < HID; ++h)
        dot += (float)hrow[h] * (float)Wcs[h];
      float f2 = (t % 2   == 0) ? 1.0f : 0.0f;
      float f3 = (t % 10  == 0) ? 1.0f : 0.0f;
      float f4 = (t % 100 == 0) ? 1.0f : 0.0f;
      dot += (float)Wcs[HID] + f2 * (float)Wcs[HID + 1]
           + f3 * (float)Wcs[HID + 2] + f4 * (float)Wcs[HID + 3];
      float co = csS[rr];                 // cs carried into this step
      if (co == 1.0f) bgtS[rr] += 1.0f;
      float cmv = cmS[rr];
      cmv = cmv * 0.1f * (1.0f - co) + dot;
      float ncs = (cmv > 0.5f) ? 1.0f : 0.0f;
      cmS[rr] = cmv;
      csS[rr] = ncs;
      // pre-compute next step's row-active flag (gate=new cs, window @ t+1)
      int sN = (t + 1) - idxS[rr];
      actS[rr] = (sN >= 0 && sN < WIN && ncs != 0.0f) ? 1.0f : 0.0f;
    }
    __syncthreads();   // new cs/act visible before next step
  }

  if (tid < RB * NCOUT) {
    int rr = tid & 15, c = tid >> 4;
    rate_out[(long long)(r0 + rr) * NCOUT + c] = sum2S[tid] / bgtS[rr];
  }
}

// ---------------------------------------------------------------------------
// Launcher. d_out = [rate (128*10 f32)] ++ [X (128*2312*300 f32)].
// ---------------------------------------------------------------------------
extern "C" void kernel_launch(void* const* d_in, const int* in_sizes, int n_in,
                              void* d_out, int out_size, void* d_ws, size_t ws_size,
                              hipStream_t stream) {
  const float* input = (const float*)d_in[0];
  const float* Xin   = (const float*)d_in[1];
  const int*   idx   = (const int*)d_in[2];
  const float* W1    = (const float*)d_in[3];
  const float* b1    = (const float*)d_in[4];
  const float* W2    = (const float*)d_in[5];
  const float* b2    = (const float*)d_in[6];
  const float* Wc    = (const float*)d_in[7];
  const float* bc    = (const float*)d_in[8];

  float*  rate = (float*)d_out;
  float*  Xout = rate + BATCH * NCOUT;
  __bf16* w1s  = (__bf16*)d_ws;

  long long totalX = (long long)BATCH * DIN * TT;
  int bpaste = (int)((totalX + 255) / 256);
  paste_kernel<<<bpaste, 256, 0, stream>>>(input, Xin, idx, Xout, totalX);

  w1_swizzle_kernel<<<(W1S_ELEMS + 255) / 256, 256, 0, stream>>>(W1, w1s);

  size_t dynLds = (size_t)AS_ELEMS * sizeof(__bf16);   // 74,752 B
  snn_scan_kernel<<<BATCH / RB, 320, dynLds, stream>>>(input, idx, w1s, b1,
                                                       W2, b2, Wc, bc, rate);
}
